// Net_4698694222696
// MI455X (gfx1250) — matrix operand order repaired and checked by reading, not compiled, not run on hardware
//
#include <hip/hip_runtime.h>

#define Gn 64
#define Nn 512
#define F_IN 7
#define HID 64
#define CLS 2
#define KP 128

typedef _Float16 half8  __attribute__((ext_vector_type(8)));
typedef _Float16 half16 __attribute__((ext_vector_type(16)));
typedef float    float8 __attribute__((ext_vector_type(8)));

#define WMMA_F16(a, b, c) \
    __builtin_amdgcn_wmma_f32_16x16x32_f16(false, (a), false, (b), (short)0, (c), false, false)

// ---- WMMA fragment loaders (f16, 16x16x32) -------------------------------
// A-matrix (16x32, MxK): lane l<16 row M=l holds K = {0..7, 16..23}+k0,
// lane l>=16 row M=l-16 holds K = {8..15, 24..31}+k0 (ISA 7.12.2 table).
__device__ __forceinline__ half16 load_a_frag(const _Float16* __restrict__ row,
                                              int k0, int hi) {
    half8 lo  = *(const half8*)(row + k0 + 8 * hi);
    half8 hi8 = *(const half8*)(row + k0 + 16 + 8 * hi);
    half16 r;
#pragma unroll
    for (int i = 0; i < 8; ++i) { r[i] = lo[i]; r[i + 8] = hi8[i]; }
    return r;
}
// B-matrix (32x16, KxN), loaded from B^T rows (row n = column n of B):
// lanes 0-15 hold K = k0+{0..7,16..23} for col n=l; lanes 16-31 hold K = k0+{8..15,24..31}.
__device__ __forceinline__ half16 load_b_frag(const _Float16* __restrict__ rowT,
                                              int k0, int hi) {
    half8 lo  = *(const half8*)(rowT + k0 + 8 * hi);
    half8 hi8 = *(const half8*)(rowT + k0 + 16 + 8 * hi);
    half16 r;
#pragma unroll
    for (int i = 0; i < 8; ++i) { r[i] = lo[i]; r[i + 8] = hi8[i]; }
    return r;
}

// ---- 2x2-blocked WMMA inner loop: 32x32 output per wave ------------------
// 8 b128 loads feed 4 wmma per k-step: 2 loads/wmma (vs 4 unblocked).
template <int LDA, int LDB, int KTOT>
__device__ __forceinline__ void gemm2x2(const _Float16* __restrict__ Abase,
                                        const _Float16* __restrict__ Bbase,
                                        int tm, int tn, int l16, int hi,
                                        float8& c00, float8& c01,
                                        float8& c10, float8& c11) {
    const _Float16* ar0 = Abase + (size_t)(tm * 32 + l16) * LDA;
    const _Float16* ar1 = Abase + (size_t)(tm * 32 + 16 + l16) * LDA;
    const _Float16* br0 = Bbase + (size_t)(tn * 32 + l16) * LDB;
    const _Float16* br1 = Bbase + (size_t)(tn * 32 + 16 + l16) * LDB;
    for (int k0 = 0; k0 < KTOT; k0 += 32) {
        half16 a0 = load_a_frag(ar0, k0, hi);
        half16 a1 = load_a_frag(ar1, k0, hi);
        half16 b0 = load_b_frag(br0, k0, hi);
        half16 b1 = load_b_frag(br1, k0, hi);
        c00 = WMMA_F16(a0, b0, c00);
        c01 = WMMA_F16(a0, b1, c01);
        c10 = WMMA_F16(a1, b0, c10);
        c11 = WMMA_F16(a1, b1, c11);
    }
}

// ---- 0: adjacency f32 -> f16 (exact: values are 0/1) ---------------------
__global__ void k_adj2h(const float* __restrict__ adj, _Float16* __restrict__ adjh) {
    int e = blockIdx.x * 256 + threadIdx.x;  // Gn*Nn*Nn = 16777216
    adjh[e] = (_Float16)adj[e];
}

// ---- 1: A2 = adj @ adj, f32 accumulate, store f16 (exact integer counts) -
__global__ void k_gemm_a2(const _Float16* __restrict__ adjh, _Float16* __restrict__ a2h) {
    int wave = (blockIdx.x << 2) | (threadIdx.x >> 5);  // Gn*16*16 waves
    int lane = threadIdx.x & 31;
    int g  = wave >> 8;
    int tm = (wave >> 4) & 15;
    int tn = wave & 15;
    int l16 = lane & 15, hi = lane >> 4;
    const _Float16* base = adjh + (size_t)g * Nn * Nn;  // symmetric: B^T rows = rows
    float8 c00 = {}, c01 = {}, c10 = {}, c11 = {};
    gemm2x2<Nn, Nn, Nn>(base, base, tm, tn, l16, hi, c00, c01, c10, c11);
    _Float16* obase = a2h + (size_t)g * Nn * Nn;
#pragma unroll
    for (int si = 0; si < 2; ++si)
#pragma unroll
        for (int sj = 0; sj < 2; ++sj) {
            const float8& c = si ? (sj ? c11 : c10) : (sj ? c01 : c00);
            int col = tn * 32 + sj * 16 + l16;
#pragma unroll
            for (int r = 0; r < 8; ++r) {
                int row = tm * 32 + si * 16 + r + 8 * hi;
                obase[(size_t)row * Nn + col] = (_Float16)c[r];
            }
        }
}

// ---- 2: M = c0*I + c1*A + c2*A2 + c3*(A2@A), A3 tile via WMMA ------------
__global__ void k_build_m(const _Float16* __restrict__ adjh, const _Float16* __restrict__ a2h,
                          const float* __restrict__ panw, float* __restrict__ Mbuf) {
    int wave = (blockIdx.x << 2) | (threadIdx.x >> 5);  // Gn*16*16 waves
    int lane = threadIdx.x & 31;
    int g  = wave >> 8;
    int tm = (wave >> 4) & 15;
    int tn = wave & 15;
    int l16 = lane & 15, hi = lane >> 4;
    float c0 = panw[0];
    float c1 = c0 * panw[1];
    float c2 = c1 * panw[2];
    float c3 = c2 * panw[3];
    size_t gb = (size_t)g * Nn * Nn;
    float8 c00 = {}, c01 = {}, c10 = {}, c11 = {};
    gemm2x2<Nn, Nn, Nn>(a2h + gb, adjh + gb, tm, tn, l16, hi, c00, c01, c10, c11);
#pragma unroll
    for (int si = 0; si < 2; ++si)
#pragma unroll
        for (int sj = 0; sj < 2; ++sj) {
            const float8& c = si ? (sj ? c11 : c10) : (sj ? c01 : c00);
            int col = tn * 32 + sj * 16 + l16;
#pragma unroll
            for (int r = 0; r < 8; ++r) {
                int row = tm * 32 + si * 16 + r + 8 * hi;
                size_t gi = gb + (size_t)row * Nn + col;
                float m = c1 * (float)adjh[gi] + c2 * (float)a2h[gi] + c3 * c[r];
                if (row == col) m += c0;
                Mbuf[gi] = m;
            }
        }
}

// ---- row reductions: mode 0 -> rsqrt(max(sum,1)) ; mode 1 -> raw sum -----
__global__ void k_rowreduce(const float* __restrict__ m, float* __restrict__ outv,
                            int ncols, int mode) {
    int row  = blockIdx.x * 8 + (threadIdx.x >> 5);
    int lane = threadIdx.x & 31;
    const float* r = m + (size_t)row * ncols;
    float s = 0.f;
    for (int j = lane; j < ncols; j += 32) s += r[j];
#pragma unroll
    for (int off = 16; off; off >>= 1) s += __shfl_xor(s, off, 32);
    if (lane == 0) outv[row] = (mode == 0) ? rsqrtf(fmaxf(s, 1.0f)) : s;
}

// ---- 3: Mn = D^-1/2 M D^-1/2 in place + f16 copy -------------------------
__global__ void k_normalize(float* __restrict__ M, _Float16* __restrict__ mnh,
                            const float* __restrict__ dvec) {
    int e = blockIdx.x * 256 + threadIdx.x;  // Gn*Nn*Nn
    int g = e >> 18;
    int i = (e >> 9) & 511;
    int j = e & 511;
    float v = M[e] * dvec[(g << 9) | i] * dvec[(g << 9) | j];
    M[e] = v;
    mnh[e] = (_Float16)v;
}

// ---- 4: xw = x @ W1, stored transposed f16 [G][HID][N] -------------------
__global__ void k_xw(const float* __restrict__ x, const float* __restrict__ w1,
                     _Float16* __restrict__ xwT) {
    int e = blockIdx.x * 256 + threadIdx.x;  // Gn*Nn*HID
    int g  = e >> 15;
    int n  = (e >> 6) & 511;
    int ho = e & 63;
    const float* xr = x + (size_t)(g * Nn + n) * F_IN;
    float s = 0.f;
#pragma unroll
    for (int f = 0; f < F_IN; ++f) s += xr[f] * w1[f * HID + ho];
    xwT[(size_t)g * HID * Nn + (size_t)ho * Nn + n] = (_Float16)s;
}

// ---- 5: h = relu(Mn @ xw + b1), WMMA over K=512, 2x2 blocked -------------
__global__ void k_h(const _Float16* __restrict__ mnh, const _Float16* __restrict__ xwT,
                    const float* __restrict__ b1, float* __restrict__ hbuf) {
    int wave = (blockIdx.x << 2) | (threadIdx.x >> 5);  // Gn*16*2 waves
    int lane = threadIdx.x & 31;
    int g  = wave >> 5;
    int tm = (wave >> 1) & 15;
    int tn = wave & 1;
    int l16 = lane & 15, hi = lane >> 4;
    float8 c00 = {}, c01 = {}, c10 = {}, c11 = {};
    gemm2x2<Nn, Nn, Nn>(mnh + (size_t)g * Nn * Nn, xwT + (size_t)g * HID * Nn,
                        tm, tn, l16, hi, c00, c01, c10, c11);
    float* obase = hbuf + (size_t)g * Nn * HID;
#pragma unroll
    for (int si = 0; si < 2; ++si)
#pragma unroll
        for (int sj = 0; sj < 2; ++sj) {
            const float8& c = si ? (sj ? c11 : c10) : (sj ? c01 : c00);
            int col = tn * 32 + sj * 16 + l16;
            float bias = b1[col];
#pragma unroll
            for (int r = 0; r < 8; ++r) {
                int row = tm * 32 + si * 16 + r + 8 * hi;
                obase[(size_t)row * HID + col] = fmaxf(c[r] + bias, 0.f);
            }
        }
}

// ---- 6: score = tanh(b0*<h,p> + b1*colsum(Mn)); exact top-k via ranks ----
__global__ void k_score_topk(const float* __restrict__ hbuf, const float* __restrict__ s2,
                             const float* __restrict__ pvec, const float* __restrict__ beta,
                             int* __restrict__ idxb, float* __restrict__ vals) {
    __shared__ float sc[Nn];
    int g = blockIdx.x;
    int n = threadIdx.x;
    const float* hr = hbuf + (size_t)(g * Nn + n) * HID;
    float s1 = 0.f;
    for (int hh = 0; hh < HID; ++hh) s1 += hr[hh] * pvec[hh];
    float s = tanhf(beta[0] * s1 + beta[1] * s2[g * Nn + n]);
    sc[n] = s;
    __syncthreads();
    int rank = 0;
    for (int j = 0; j < Nn; ++j) {
        float sj = sc[j];
        rank += (sj > s) || (sj == s && j < n);
    }
    if (rank < KP) { idxb[g * KP + rank] = n; vals[g * KP + rank] = s; }
}

// ---- 7: xp = h[idx] * vals ----------------------------------------------
__global__ void k_pool(const float* __restrict__ hbuf, const int* __restrict__ idxb,
                       const float* __restrict__ vals, float* __restrict__ xp) {
    int e = blockIdx.x * 256 + threadIdx.x;  // Gn*KP*HID
    int g  = e >> 13;
    int k  = (e >> 6) & 127;
    int ho = e & 63;
    int node = idxb[(g << 7) + k];
    xp[e] = hbuf[(size_t)(g * Nn + node) * HID + ho] * vals[(g << 7) + k];
}

// ---- 8: Mp = Mn[idx][:,idx] ---------------------------------------------
__global__ void k_mp(const float* __restrict__ Mbuf, const int* __restrict__ idxb,
                     float* __restrict__ Mp) {
    int e = blockIdx.x * 256 + threadIdx.x;  // Gn*KP*KP
    int g = e >> 14;
    int i = (e >> 7) & 127;
    int j = e & 127;
    int ni = idxb[(g << 7) + i];
    int nj = idxb[(g << 7) + j];
    Mp[e] = Mbuf[(size_t)g * Nn * Nn + (size_t)ni * Nn + nj];
}

// ---- 9: GCN degrees of (Mp + I) -----------------------------------------
__global__ void k_gcn_deg(const float* __restrict__ Mp, float* __restrict__ di) {
    int row  = blockIdx.x * 8 + (threadIdx.x >> 5);  // Gn*KP rows
    int lane = threadIdx.x & 31;
    int i = row & (KP - 1);
    const float* r = Mp + (size_t)row * KP;
    float s = 0.f;
    for (int j = lane; j < KP; j += 32) s += r[j] + ((j == i) ? 1.0f : 0.0f);
#pragma unroll
    for (int off = 16; off; off >>= 1) s += __shfl_xor(s, off, 32);
    if (lane == 0) di[row] = (s > 0.f) ? rsqrtf(s) : 0.f;
}

// ---- 10: An = D^-1/2 (Mp+I) D^-1/2 -> f16 --------------------------------
__global__ void k_an(const float* __restrict__ Mp, const float* __restrict__ di,
                     _Float16* __restrict__ anh) {
    int e = blockIdx.x * 256 + threadIdx.x;  // Gn*KP*KP
    int g = e >> 14;
    int i = (e >> 7) & 127;
    int j = e & 127;
    float v = (Mp[e] + ((i == j) ? 1.f : 0.f)) * di[(g << 7) | i] * di[(g << 7) | j];
    anh[e] = (_Float16)v;
}

// ---- 11: xg = xp @ gcn_w, stored transposed f16 [G][HID][KP] -------------
__global__ void k_xg(const float* __restrict__ xp, const float* __restrict__ gcnw,
                     _Float16* __restrict__ xgT) {
    int e = blockIdx.x * 256 + threadIdx.x;  // Gn*KP*HID
    int g  = e >> 13;
    int k  = (e >> 6) & 127;
    int ho = e & 63;
    const float* xr = xp + (size_t)((g << 7) + k) * HID;
    float s = 0.f;
    for (int hh = 0; hh < HID; ++hh) s += xr[hh] * gcnw[hh * HID + ho];
    xgT[(size_t)g * HID * KP + (size_t)ho * KP + k] = (_Float16)s;
}

// ---- 12: h2 = relu(An @ xg + gcn_b), WMMA over K=128, 2x2 blocked --------
__global__ void k_h2(const _Float16* __restrict__ anh, const _Float16* __restrict__ xgT,
                     const float* __restrict__ gcnb, float* __restrict__ h2) {
    int wave = (blockIdx.x << 2) | (threadIdx.x >> 5);  // Gn*4*2 waves
    int lane = threadIdx.x & 31;
    int g  = wave >> 3;
    int tm = (wave >> 1) & 3;
    int tn = wave & 1;
    int l16 = lane & 15, hi = lane >> 4;
    float8 c00 = {}, c01 = {}, c10 = {}, c11 = {};
    gemm2x2<KP, KP, KP>(anh + (size_t)g * KP * KP, xgT + (size_t)g * HID * KP,
                        tm, tn, l16, hi, c00, c01, c10, c11);
    float* obase = h2 + (size_t)g * KP * HID;
#pragma unroll
    for (int si = 0; si < 2; ++si)
#pragma unroll
        for (int sj = 0; sj < 2; ++sj) {
            const float8& c = si ? (sj ? c11 : c10) : (sj ? c01 : c00);
            int col = tn * 32 + sj * 16 + l16;
            float bias = gcnb[col];
#pragma unroll
            for (int r = 0; r < 8; ++r) {
                int row = tm * 32 + si * 16 + r + 8 * hi;
                obase[(size_t)row * HID + col] = fmaxf(c[r] + bias, 0.f);
            }
        }
}

// ---- 13: sum pool + linear + log_softmax ---------------------------------
__global__ void k_head(const float* __restrict__ h2, const float* __restrict__ linw,
                       const float* __restrict__ linb, float* __restrict__ out) {
    __shared__ float pooled[HID];
    __shared__ float lg[CLS];
    int g = blockIdx.x;
    int t = threadIdx.x;  // 64
    const float* hb = h2 + (size_t)g * KP * HID;
    float s = 0.f;
    for (int k = 0; k < KP; ++k) s += hb[k * HID + t];
    pooled[t] = s;
    __syncthreads();
    if (t < CLS) {
        float l = linb[t];
        for (int hh = 0; hh < HID; ++hh) l += pooled[hh] * linw[hh * CLS + t];
        lg[t] = l;
    }
    __syncthreads();
    if (t < CLS) {
        float m = fmaxf(lg[0], lg[1]);
        float lse = m + logf(expf(lg[0] - m) + expf(lg[1] - m));
        out[g * CLS + t] = lg[t] - lse;
    }
}

extern "C" void kernel_launch(void* const* d_in, const int* in_sizes, int n_in,
                              void* d_out, int out_size, void* d_ws, size_t ws_size,
                              hipStream_t stream) {
    const float* x    = (const float*)d_in[0];
    const float* adj  = (const float*)d_in[1];
    const float* panw = (const float*)d_in[2];
    const float* w1   = (const float*)d_in[3];
    const float* b1   = (const float*)d_in[4];
    const float* pvec = (const float*)d_in[5];
    const float* beta = (const float*)d_in[6];
    const float* gcnw = (const float*)d_in[7];
    const float* gcnb = (const float*)d_in[8];
    const float* linw = (const float*)d_in[9];
    const float* linb = (const float*)d_in[10];
    float* out = (float*)d_out;

    // Workspace layout (128 MB total, with reuse):
    //   [0, 64MB)      Mbuf f32 (M, then Mn in place)
    //   [64, 96MB)     adjh f16  -> reused as mnh f16 after k_build_m
    //   [96, 128MB)    a2h f16   -> reused for all small buffers after k_build_m
    char* ws = (char*)d_ws;
    float*    Mbuf = (float*)ws;
    _Float16* adjh = (_Float16*)(ws + 67108864);
    _Float16* mnh  = adjh;  // reuse: adjh dead after k_build_m
    char* r2 = ws + 100663296;
    _Float16* a2h  = (_Float16*)r2;  // dead after k_build_m
    float*    dvec = (float*)(r2 + 0);
    float*    s2   = (float*)(r2 + 131072);
    _Float16* xwT  = (_Float16*)(r2 + 262144);
    float*    hbuf = (float*)(r2 + 4456448);
    float*    vals = (float*)(r2 + 12845056);
    int*      idxb = (int*)(r2 + 12877824);
    float*    xp   = (float*)(r2 + 12910592);
    float*    Mp   = (float*)(r2 + 15007744);
    float*    di   = (float*)(r2 + 19202048);
    _Float16* anh  = (_Float16*)(r2 + 19234816);
    _Float16* xgT  = (_Float16*)(r2 + 21331968);
    float*    h2   = (float*)(r2 + 22380544);

    k_adj2h<<<65536, 256, 0, stream>>>(adj, adjh);
    k_gemm_a2<<<4096, 128, 0, stream>>>(adjh, a2h);            // 16384 waves, 32x32/wave
    k_build_m<<<4096, 128, 0, stream>>>(adjh, a2h, panw, Mbuf);
    k_rowreduce<<<4096, 256, 0, stream>>>(Mbuf, dvec, Nn, 0);  // deg -> rsqrt
    k_normalize<<<65536, 256, 0, stream>>>(Mbuf, mnh, dvec);
    k_rowreduce<<<4096, 256, 0, stream>>>(Mbuf, s2, Nn, 1);    // colsum(Mn) (symmetric)
    k_xw<<<8192, 256, 0, stream>>>(x, w1, xwT);
    k_h<<<512, 128, 0, stream>>>(mnh, xwT, b1, hbuf);          // 2048 waves
    k_score_topk<<<Gn, Nn, 0, stream>>>(hbuf, s2, pvec, beta, idxb, vals);
    k_pool<<<2048, 256, 0, stream>>>(hbuf, idxb, vals, xp);
    k_mp<<<4096, 256, 0, stream>>>(Mbuf, idxb, Mp);
    k_gcn_deg<<<1024, 256, 0, stream>>>(Mp, di);
    k_an<<<4096, 256, 0, stream>>>(Mp, di, anh);
    k_xg<<<2048, 256, 0, stream>>>(xp, gcnw, xgT);
    k_h2<<<128, 128, 0, stream>>>(anh, xgT, gcnb, h2);         // 512 waves
    k_head<<<Gn, HID, 0, stream>>>(h2, linw, linb, out);
}
